// IORMask2Points_48541720379605
// MI455X (gfx1250) — matrix operand order repaired and checked by reading, not compiled, not run on hardware
//
#include <hip/hip_runtime.h>
#include <hip/hip_bf16.h>
#include <stdint.h>

// Problem dims (fixed by the reference setup_inputs)
#define B_   64
#define N_   4
#define H_   512
#define W_   512
#define HW_  (H_ * W_)          // 262144 pixels per batch item
#define NUM_ 1024               // points per batch item
#define WORDS_PER_B (HW_ / 32)  // 8192 bitmask words per batch
#define CHUNK_PX 1024           // pixels per chunk
#define CHUNKS_PER_B (HW_ / CHUNK_PX)  // 256 chunks per batch
#define WORDS_PER_CHUNK (CHUNK_PX / 32) // 32 words per chunk

typedef __attribute__((ext_vector_type(16))) _Float16 v16h;
typedef __attribute__((ext_vector_type(8)))  float    v8f;

// ---------------------------------------------------------------------------
// Kernel 1: stream 256 MiB of masks once. Each thread owns one 32-pixel word:
// loads 32 floats from each of the 4 N-planes (float4 = GLOBAL_LOAD_B128),
// thresholds the sum, packs a 32-bit validity word, popcounts, and a wave32
// shfl_xor reduction produces the per-chunk (1024 px) valid count.
// ---------------------------------------------------------------------------
__global__ __launch_bounds__(256)
void iorm_bitmask_count(const float* __restrict__ masks,
                        uint32_t* __restrict__ words,
                        int* __restrict__ counts) {
    const int wg = blockIdx.x * blockDim.x + threadIdx.x;   // global word id
    const int b    = wg >> 13;          // / WORDS_PER_B
    const int widx = wg & (WORDS_PER_B - 1);
    const size_t plane = (size_t)HW_;
    const float* p = masks + (size_t)b * N_ * plane + (size_t)widx * 32;

    uint32_t bits = 0;
#pragma unroll
    for (int i = 0; i < 8; ++i) {
        float4 a = *(const float4*)(p + i * 4);
        float4 c = *(const float4*)(p + plane + i * 4);
        float4 d = *(const float4*)(p + 2 * plane + i * 4);
        float4 e = *(const float4*)(p + 3 * plane + i * 4);
        float sx = a.x + c.x + d.x + e.x;
        float sy = a.y + c.y + d.y + e.y;
        float sz = a.z + c.z + d.z + e.z;
        float sw = a.w + c.w + d.w + e.w;
        uint32_t nib = (uint32_t)(sx > 0.5f)
                     | ((uint32_t)(sy > 0.5f) << 1)
                     | ((uint32_t)(sz > 0.5f) << 2)
                     | ((uint32_t)(sw > 0.5f) << 3);
        bits |= nib << (i * 4);
    }
    words[wg] = bits;

    int pc = __popc(bits);
    // wave32 reduction: 32 consecutive threads == one chunk (32 words)
#pragma unroll
    for (int off = 16; off > 0; off >>= 1)
        pc += __shfl_xor(pc, off, 32);
    if ((threadIdx.x & 31) == 0)
        counts[wg >> 5] = pc;
}

// ---------------------------------------------------------------------------
// Kernel 2: exact 256-element inclusive scan per batch using WMMA.
// counts (<=1024, exact in f16) viewed as 16x16 matrix M (row-major).
//   C1 = M x U        (U = upper-tri ones, incl diag)  -> within-row prefix
//   C2 = Lstrict x M  (L = strict lower-tri ones)      -> col-sums of prev rows
//   P[row] = sum over 16 cols of C2[row]  (half-wave shfl reduction)
//   csum[row*16+col] = C1[row][col] + P[row]           (exact, < 2^24)
// One wave (32 lanes) per batch item; EXEC is all-ones throughout (no
// divergence), as WMMA requires. Register layouts follow the CDNA5 ISA
// 16-bit A / B and f32 C/D tables.
// ---------------------------------------------------------------------------
__global__ __launch_bounds__(32)
void iorm_scan_wmma(const int* __restrict__ counts,
                    int* __restrict__ csum) {
    const int b    = blockIdx.x;
    const int lane = threadIdx.x;        // 0..31
    const int hi   = (lane >= 16) ? 1 : 0;
    const int row  = lane & 15;
    const int* c = counts + b * CHUNKS_PER_B;

    // ---- A operand: M.  lanes<16 hold K=0..7 of row `lane`; lanes>=16 hold
    // K=8..15 of row `lane-16`; halves 8..15 are K=16..31 padding (zero).
    v16h aM;
#pragma unroll
    for (int k = 0; k < 8; ++k)
        aM[k] = (_Float16)(float)c[row * 16 + (hi ? 8 : 0) + k];
#pragma unroll
    for (int k = 8; k < 16; ++k) aM[k] = (_Float16)0.0f;

    // ---- B operand: U (incl upper-tri).  halves 0..7 hold K=8*hi+k: (K<=n)?1:0 with
    // n=lane&15; halves 8..15 are K=16..31 padding (zero).
    v16h bU;
#pragma unroll
    for (int k = 0; k < 16; ++k)
        bU[k] = (_Float16)((k < 8 && hi * 8 + k <= row) ? 1.0f : 0.0f);

    v8f c1 = {};
    c1 = __builtin_amdgcn_wmma_f32_16x16x32_f16(false, aM, false, bU,
                                                (short)0, c1, false, false);

    // ---- A operand: Lstrict.  a[row][k] = (k < row).
    v16h aL;
#pragma unroll
    for (int k = 0; k < 8; ++k) {
        int kk = (hi ? 8 : 0) + k;
        aL[k] = (_Float16)((kk < row) ? 1.0f : 0.0f);
    }
#pragma unroll
    for (int k = 8; k < 16; ++k) aL[k] = (_Float16)0.0f;

    // ---- B operand: M in B-layout.  halves 0..7: K=8*hi+k, M[K][n] = c[K*16+n]; 8..15 zero.
    v16h bM;
#pragma unroll
    for (int k = 0; k < 16; ++k)
        bM[k] = (_Float16)((k < 8) ? (float)c[(hi * 8 + k) * 16 + row] : 0.0f);

    v8f c2 = {};
    c2 = __builtin_amdgcn_wmma_f32_16x16x32_f16(false, aL, false, bM,
                                                (short)0, c2, false, false);

    // Row offsets: reduce each C2 register across its 16-lane half
    // (xor masks < 16 never cross the half-wave boundary).
    float p[8];
#pragma unroll
    for (int g = 0; g < 8; ++g) {
        float v = c2[g];
#pragma unroll
        for (int off = 8; off > 0; off >>= 1)
            v += __shfl_xor(v, off, 32);
        p[g] = v;
    }

    int* out = csum + b * CHUNKS_PER_B;
#pragma unroll
    for (int g = 0; g < 8; ++g) {
        int r = g + (hi ? 8 : 0);       // C/D layout: VGPR g -> M=g (+8 hi)
        float val = c1[g] + p[g];       // exact integer < 2^24
        out[r * 16 + row] = (int)(val + 0.5f);
    }
}

// ---------------------------------------------------------------------------
// Kernel 3: inverse-CDF lookup. One thread per (batch, point): compute rank,
// binary-search the 256-entry chunk cumsum (L2-resident), then popcount-scan
// <=32 bitmask words and select the exact bit.
// ---------------------------------------------------------------------------
__global__ __launch_bounds__(256)
void iorm_select(const float* __restrict__ rand_u,
                 const int* __restrict__ csum,
                 const uint32_t* __restrict__ words,
                 float* __restrict__ out, int out_size) {
    const int gid = blockIdx.x * blockDim.x + threadIdx.x;  // 0..65535
    const int b = gid >> 10;
    const int j = gid & (NUM_ - 1);

    const int* cs = csum + b * CHUNKS_PER_B;
    const int total = cs[CHUNKS_PER_B - 1];
    const int cnt = max(total, 1);

    float u = rand_u[b * NUM_ + j];
    int r = (int)(u * (float)cnt);          // trunc toward zero, matches jnp
    r = min(r, cnt - 1);
    const int t = r + 1;

    int pos;
    if (total < t) {
        pos = HW_;                          // empty mask: searchsorted -> end
    } else {
        int lo = 0, hi = CHUNKS_PER_B;      // first chunk with cs[c] >= t
        while (lo < hi) {
            int mid = (lo + hi) >> 1;
            if (cs[mid] >= t) hi = mid; else lo = mid + 1;
        }
        const int chunk = lo;
        const int base = chunk ? cs[chunk - 1] : 0;
        int rem = t - base;                 // 1-indexed rank within chunk
        const uint32_t* wp = words + b * WORDS_PER_B + chunk * WORDS_PER_CHUNK;
        int w = 0;
        uint32_t word = wp[0];
        int pc = __popc(word);
        while (rem > pc && w < WORDS_PER_CHUNK - 1) {
            rem -= pc;
            word = wp[++w];
            pc = __popc(word);
        }
        // rem-th (1-indexed) set bit of `word`
        for (int i = 1; i < rem; ++i) word &= word - 1;
        int bit = __ffs(word) - 1;
        pos = chunk * CHUNK_PX + w * 32 + bit;
    }

    out[gid * 2 + 0] = (float)(pos >> 9);   // idx_h = pos / W
    out[gid * 2 + 1] = (float)(pos & 511);  // idx_w = pos % W

    // Tail of d_out = the (H, W) tuple; both are 512.
    if (gid == 0) {
        for (int k = B_ * NUM_ * 2; k < out_size; ++k)
            out[k] = 512.0f;
    }
}

// ---------------------------------------------------------------------------
extern "C" void kernel_launch(void* const* d_in, const int* in_sizes, int n_in,
                              void* d_out, int out_size, void* d_ws, size_t ws_size,
                              hipStream_t stream) {
    const float* masks  = (const float*)d_in[0];   // [64,4,512,512] f32
    const float* rand_u = (const float*)d_in[1];   // [64,1024] f32
    (void)in_sizes; (void)n_in; (void)ws_size;

    // Workspace layout (all overwritten every call; deterministic):
    //   [0, 2 MiB)         : validity bitmask words  (B * 8192 u32)
    //   [2 MiB, +64 KiB)   : per-chunk counts        (B * 256 i32)
    //   [.., +64 KiB)      : per-chunk inclusive sum (B * 256 i32)
    uint32_t* words  = (uint32_t*)d_ws;
    int*      counts = (int*)((char*)d_ws + (size_t)B_ * WORDS_PER_B * 4);
    int*      csum   = (int*)((char*)counts + (size_t)B_ * CHUNKS_PER_B * 4);

    const int total_words = B_ * WORDS_PER_B;      // 524288
    iorm_bitmask_count<<<total_words / 256, 256, 0, stream>>>(masks, words, counts);

    iorm_scan_wmma<<<B_, 32, 0, stream>>>(counts, csum);

    iorm_select<<<(B_ * NUM_) / 256, 256, 0, stream>>>(rand_u, csum, words,
                                                       (float*)d_out, out_size);
}
